// MemoryManager_31181462569520
// MI455X (gfx1250) — matrix operand order repaired and checked by reading, not compiled, not run on hardware
//
#include <hip/hip_runtime.h>

// Problem constants from the reference (r=2048 is fixed by setup_inputs).
#define B_    4
#define T_    8192
#define C_    1024
#define H_    16
#define D_    64      // metric dim = C/H
#define TH_   4096    // T/2 (even / odd token counts)
#define R_    2048    // merged count
#define OUTT_ 6144    // TH - R + TH

typedef __attribute__((ext_vector_type(16))) _Float16 v16h;
typedef __attribute__((ext_vector_type(8)))  _Float16 v8h;
typedef __attribute__((ext_vector_type(8)))  float    v8f;

// ---------------------------------------------------------------------------
// K1: metric = normalize(mean over heads of x), stored f16, split even/odd.
// One wave32 per token; lane l owns dims {l, l+32}.
// ---------------------------------------------------------------------------
__global__ __launch_bounds__(256) void metric_kernel(const float* __restrict__ x,
                                                     _Float16* __restrict__ ma,
                                                     _Float16* __restrict__ mb) {
  int token = blockIdx.x * 8 + (threadIdx.x >> 5);   // [0, B*T)
  int lane  = threadIdx.x & 31;
  const float* row = x + (size_t)token * C_;
  float m0 = 0.f, m1 = 0.f;
#pragma unroll
  for (int h = 0; h < H_; ++h) {
    m0 += row[h * D_ + lane];
    m1 += row[h * D_ + lane + 32];
  }
  m0 *= (1.f / H_); m1 *= (1.f / H_);
  float ss = m0 * m0 + m1 * m1;
#pragma unroll
  for (int m = 16; m >= 1; m >>= 1) ss += __shfl_xor(ss, m, 32);
  float rn = rsqrtf(ss);
  // even tokens -> ma row token/2 ; odd tokens -> mb row token/2
  _Float16* dst = ((token & 1) ? mb : ma) + (size_t)(token >> 1) * D_;
  dst[lane]      = (_Float16)(m0 * rn);
  dst[lane + 32] = (_Float16)(m1 * rn);
}

// ---------------------------------------------------------------------------
// K2: scores = ma . mb^T via v_wmma_f32_16x16x32_f16 (K=64 -> 2 chained WMMAs),
// fused rowwise max/argmax epilogue (scores never hit memory).
// Block = 8 waves; each wave owns the block's 16-row M tile and strides N tiles.
// N-sweep unrolled 2x: 8 loads in flight feeding 4 WMMAs on 2 accumulators.
// ---------------------------------------------------------------------------
__global__ __launch_bounds__(256) void scores_max_kernel(const _Float16* __restrict__ ma,
                                                         const _Float16* __restrict__ mb,
                                                         float* __restrict__ node_max,
                                                         int*   __restrict__ node_idx) {
  __shared__ float sval[8][16];
  __shared__ int   sidx[8][16];

  const int lane = threadIdx.x & 31;
  const int wave = threadIdx.x >> 5;     // 0..7
  const int l16  = lane & 15;
  const int g    = lane >> 4;            // half-wave selector
  const int mt   = blockIdx.x & 255;     // M tile (0..255)
  const int b    = blockIdx.x >> 8;      // batch

  // A fragment (16x64 f16, reused across all N tiles).
  // A layout (contiguous, matches B): lanes row M=l16 hold K {g*16..g*16+15}.
  const _Float16* Abase = ma + ((size_t)b * TH_ + mt * 16 + l16) * D_;
  union { v16h v; v8h h[2]; } ua0, ua1;
  ua0.h[0] = *(const v8h*)(Abase + g * 16);
  ua0.h[1] = *(const v8h*)(Abase + g * 16 + 8);
  ua1.h[0] = *(const v8h*)(Abase + 32 + g * 16);
  ua1.h[1] = *(const v8h*)(Abase + 32 + g * 16 + 8);

  float best[8];
  int   bidx[8];
#pragma unroll
  for (int v = 0; v < 8; ++v) { best[v] = -__builtin_inff(); bidx[v] = 0; }

  // Per-wave N stripe: tiles wave, wave+8, ..., processed two per iteration.
  const _Float16* Bstripe = mb + ((size_t)b * TH_ + wave * 16 + l16) * D_;
  for (int it = 0; it < (TH_ / 16) / 16; ++it) {
    const int nt0 = wave + 16 * it;          // first tile of the pair
    const _Float16* B0 = Bstripe + (size_t)(16 * it) * 16 * D_;
    const _Float16* B1 = B0 + (size_t)8 * 16 * D_;   // tile nt0+8

    // Issue all 8 fragment loads before any WMMA (max loads in flight).
    union { v16h v; v8h h[2]; } ub0a, ub1a, ub0b, ub1b;
    ub0a.h[0] = *(const v8h*)(B0 + g * 16);
    ub0a.h[1] = *(const v8h*)(B0 + g * 16 + 8);
    ub1a.h[0] = *(const v8h*)(B0 + 32 + g * 16);
    ub1a.h[1] = *(const v8h*)(B0 + 32 + g * 16 + 8);
    ub0b.h[0] = *(const v8h*)(B1 + g * 16);
    ub0b.h[1] = *(const v8h*)(B1 + g * 16 + 8);
    ub1b.h[0] = *(const v8h*)(B1 + 32 + g * 16);
    ub1b.h[1] = *(const v8h*)(B1 + 32 + g * 16 + 8);

    // Prefetch next iteration's B rows (global_prefetch_b8); stream is tiny
    // (2 MB) so this keeps it WGP$/L2 resident.
    __builtin_prefetch(B0 + (size_t)16 * 16 * D_, 0, 3);
    __builtin_prefetch(B1 + (size_t)16 * 16 * D_, 0, 3);

    v8f c0 = {}, c1 = {};
    c0 = __builtin_amdgcn_wmma_f32_16x16x32_f16(false, ua0.v, false, ub0a.v,
                                                (short)0, c0, false, false);
    c0 = __builtin_amdgcn_wmma_f32_16x16x32_f16(false, ua1.v, false, ub1a.v,
                                                (short)0, c0, false, false);
    c1 = __builtin_amdgcn_wmma_f32_16x16x32_f16(false, ua0.v, false, ub0b.v,
                                                (short)0, c1, false, false);
    c1 = __builtin_amdgcn_wmma_f32_16x16x32_f16(false, ua1.v, false, ub1b.v,
                                                (short)0, c1, false, false);

    const int N0 = nt0 * 16 + l16;        // this lane's column in tile 0
    const int N1 = N0 + 8 * 16;           // and in tile 1
#pragma unroll
    for (int v = 0; v < 8; ++v) {         // VGPR v -> row M = v + 8*g
      float v0 = c0[v];
      if (v0 > best[v]) { best[v] = v0; bidx[v] = N0; }  // N ascending => first max kept
      float v1 = c1[v];
      if (v1 > best[v]) { best[v] = v1; bidx[v] = N1; }
    }
  }

  // Reduce over the 16 lanes of each half-wave (columns), first-occurrence ties.
#pragma unroll
  for (int v = 0; v < 8; ++v) {
    float bv = best[v]; int bi = bidx[v];
#pragma unroll
    for (int m = 1; m < 16; m <<= 1) {
      float ov = __shfl_xor(bv, m, 32);
      int   oi = __shfl_xor(bi, m, 32);
      if (ov > bv || (ov == bv && oi < bi)) { bv = ov; bi = oi; }
    }
    best[v] = bv; bidx[v] = bi;
  }

  if (l16 == 0) {
#pragma unroll
    for (int v = 0; v < 8; ++v) { sval[wave][g * 8 + v] = best[v]; sidx[wave][g * 8 + v] = bidx[v]; }
  }
  __syncthreads();

  if (threadIdx.x < 16) {
    int row = threadIdx.x;
    float bv = sval[0][row]; int bi = sidx[0][row];
#pragma unroll
    for (int w = 1; w < 8; ++w) {
      float ov = sval[w][row]; int oi = sidx[w][row];
      if (ov > bv || (ov == bv && oi < bi)) { bv = ov; bi = oi; }
    }
    node_max[b * TH_ + mt * 16 + row] = bv;
    node_idx[b * TH_ + mt * 16 + row] = bi;
  }
}

// ---------------------------------------------------------------------------
// K3: per-batch bitonic argsort of 4096 keys in LDS, key = (value desc, idx asc)
// which reproduces a stable descending argsort (matches jnp.argsort(-v)).
// ---------------------------------------------------------------------------
__global__ __launch_bounds__(1024) void sort_kernel(const float* __restrict__ node_max,
                                                    int* __restrict__ edge_idx) {
  __shared__ float sv[TH_];
  __shared__ int   si[TH_];
  const int b = blockIdx.x;
  for (int i = threadIdx.x; i < TH_; i += 1024) { sv[i] = node_max[b * TH_ + i]; si[i] = i; }
  __syncthreads();

  for (int k = 2; k <= TH_; k <<= 1) {
    for (int j = k >> 1; j > 0; j >>= 1) {
      for (int t = threadIdx.x; t < TH_ / 2; t += 1024) {
        int i = ((t & ~(j - 1)) << 1) | (t & (j - 1));
        int p = i | j;
        bool up = ((i & k) == 0);
        float av = sv[i], bv = sv[p];
        int   ai = si[i], bi = si[p];
        bool aBefore = (av > bv) || (av == bv && ai < bi);  // custom "<"
        if (up ? !aBefore : aBefore) {
          sv[i] = bv; sv[p] = av; si[i] = bi; si[p] = ai;
        }
      }
      __syncthreads();
    }
  }
  for (int i = threadIdx.x; i < TH_; i += 1024) edge_idx[b * TH_ + i] = si[i];
}

// K3b: dst_idx[s] = node_idx[src_idx[s]] (src_idx = first R entries of edge_idx)
__global__ __launch_bounds__(256) void dst_kernel(const int* __restrict__ edge_idx,
                                                  const int* __restrict__ node_idx,
                                                  int* __restrict__ dst_idx) {
  int s = blockIdx.x * 256 + threadIdx.x;
  int b = blockIdx.y;
  dst_idx[b * R_ + s] = node_idx[b * TH_ + edge_idx[b * TH_ + s]];
}

// ---------------------------------------------------------------------------
// K4: unmerged gather: out[b, p, :] = x[b, 2*edge_idx[R+p], :]  (size==1 so /1)
// ---------------------------------------------------------------------------
__global__ __launch_bounds__(256) void gather_unm_kernel(const float* __restrict__ x,
                                                         const int* __restrict__ edge_idx,
                                                         float* __restrict__ out) {
  int p = blockIdx.x, b = blockIdx.y;
  int e = edge_idx[b * TH_ + R_ + p];
  const float4* srow = (const float4*)(x + ((size_t)(b * T_) + 2 * e) * C_);
  float4* orow = (float4*)(out + ((size_t)(b * OUTT_) + p) * C_);
  orow[threadIdx.x] = srow[threadIdx.x];
}

// ---------------------------------------------------------------------------
// K5: dst rows: out[b, R+j, :] = (x_odd[j] + sum_{s: dst[s]==j} x_even[src[s]])
//                                / (1 + count). Deterministic scan of the tiny
//  (L2-resident) dst list instead of float atomics.
// ---------------------------------------------------------------------------
__global__ __launch_bounds__(256) void merge_dst_kernel(const float* __restrict__ x,
                                                        const int* __restrict__ edge_idx,
                                                        const int* __restrict__ dst_idx,
                                                        float* __restrict__ out) {
  int j = blockIdx.x, b = blockIdx.y;
  const float4* xrow = (const float4*)(x + ((size_t)(b * T_) + 2 * j + 1) * C_);
  float4 acc = xrow[threadIdx.x];
  float cnt = 1.f;
  for (int s = 0; s < R_; ++s) {
    if (dst_idx[b * R_ + s] == j) {
      int e = edge_idx[b * TH_ + s];
      const float4* srow = (const float4*)(x + ((size_t)(b * T_) + 2 * e) * C_);
      float4 v = srow[threadIdx.x];
      acc.x += v.x; acc.y += v.y; acc.z += v.z; acc.w += v.w;
      cnt += 1.f;
    }
  }
  float inv = 1.f / cnt;
  acc.x *= inv; acc.y *= inv; acc.z *= inv; acc.w *= inv;
  float4* orow = (float4*)(out + ((size_t)(b * OUTT_) + R_ + j) * C_);
  orow[threadIdx.x] = acc;
}

// ---------------------------------------------------------------------------
extern "C" void kernel_launch(void* const* d_in, const int* in_sizes, int n_in,
                              void* d_out, int out_size, void* d_ws, size_t ws_size,
                              hipStream_t stream) {
  (void)in_sizes; (void)n_in; (void)out_size; (void)ws_size;
  const float* x = (const float*)d_in[0];   // (4, 8192, 1024) f32
  // d_in[1] is r (==2048, compile-time constant in this problem)
  float* out = (float*)d_out;               // (4, 6144, 1024) f32

  char* ws = (char*)d_ws;
  _Float16* ma  = (_Float16*)(ws + 0);          // B*TH*D f16 = 2 MB
  _Float16* mb  = (_Float16*)(ws + (2u << 20)); // 2 MB
  float* nmax   = (float*)(ws + (4u << 20));                 // 64 KB
  int*   nidx   = (int*)  (ws + (4u << 20) + 1 * 65536u);    // 64 KB
  int*   eidx   = (int*)  (ws + (4u << 20) + 2 * 65536u);    // 64 KB
  int*   didx   = (int*)  (ws + (4u << 20) + 3 * 65536u);    // 32 KB

  metric_kernel    <<<dim3(B_ * T_ / 8),   dim3(256),  0, stream>>>(x, ma, mb);
  scores_max_kernel<<<dim3(B_ * (TH_/16)), dim3(256),  0, stream>>>(ma, mb, nmax, nidx);
  sort_kernel      <<<dim3(B_),            dim3(1024), 0, stream>>>(nmax, eidx);
  dst_kernel       <<<dim3(R_ / 256, B_),  dim3(256),  0, stream>>>(eidx, nidx, didx);
  gather_unm_kernel<<<dim3(R_, B_),        dim3(256),  0, stream>>>(x, eidx, out);
  merge_dst_kernel <<<dim3(TH_, B_),       dim3(256),  0, stream>>>(x, eidx, didx, out);
}
